// GCN_dev_86388972192142
// MI455X (gfx1250) — compile-verified, matrix operand order checked by reading, not run on hardware
//
#include <hip/hip_runtime.h>
#include <math.h>

#define NN   50000
#define NE   800000
#define IND  128
#define HID  64
#define HEADS 4
#define CLS  2
#define HO3  (HEADS * HID)   // 256
#define HO4  (HEADS * CLS)   // 8
#define NEG_SLOPE 0.2f

typedef __attribute__((ext_vector_type(16))) __bf16 v16bf;
typedef __attribute__((ext_vector_type(8)))  float  v8f;

union BF16Frag { v16bf v; uint4 q[2]; };

__device__ __forceinline__ unsigned short f2bf(float f) {
  unsigned int u = __float_as_uint(f);
  unsigned int r = (u + 0x7FFFu + ((u >> 16) & 1u)) >> 16;
  return (unsigned short)r;
}

// float atomic max via signed/unsigned integer atomics (works with -inf init)
__device__ __forceinline__ void atomicMaxF(float* addr, float val) {
  if (val >= 0.0f) atomicMax((int*)addr, __float_as_int(val));
  else             atomicMin((unsigned int*)addr, (unsigned int)__float_as_int(val));
}

// ---------------- fp32 -> bf16 convert ----------------
__global__ void k_f2bf(const float* __restrict__ in, unsigned short* __restrict__ out, int n) {
  int i = blockIdx.x * blockDim.x + threadIdx.x;
  if (i < n) out[i] = f2bf(in[i]);
}

// ---------------- WMMA NT GEMM: C[M,Nc] = A[M,K] * W[Nc,K]^T ----------------
// bf16 in, f32 accumulate. One 16x64 tile per wave32 (A fragment reused over
// 4 B fragments -> 4 v_wmma per 10 b128 loads). K % 32 == 0, Nc % 64 == 0.
__global__ void k_gemm_bf16_nt(const unsigned short* __restrict__ A,
                               const unsigned short* __restrict__ Bw,
                               float* __restrict__ C,
                               int M, int Nc, int K) {
  const int lane   = threadIdx.x & 31;
  const int ngroups = Nc >> 6;          // 64-wide column groups
  const int mtiles  = M >> 4;
  const int wid = blockIdx.x * (blockDim.x >> 5) + (threadIdx.x >> 5);
  if (wid >= mtiles * ngroups) return;
  const int mt = wid / ngroups;
  const int ng = wid - mt * ngroups;

  const int lmod = lane & 15;
  const int klo  = (lane >> 4) << 3;    // 0 or 8 per documented 16-bit A/B layout
  const unsigned short* a_ptr = A  + (size_t)(mt * 16 + lmod) * K + klo;
  const unsigned short* b_ptr = Bw + (size_t)(ng * 64 + lmod) * K + klo;
  const size_t bstep = (size_t)16 * K;

  v8f acc0 = {}, acc1 = {}, acc2 = {}, acc3 = {};
  for (int k = 0; k < K; k += 32) {
    BF16Frag a, b0, b1, b2, b3;
    a.q[0]  = *(const uint4*)(a_ptr + k);              // K = klo..klo+7
    a.q[1]  = *(const uint4*)(a_ptr + k + 16);         // K = 16+klo..
    b0.q[0] = *(const uint4*)(b_ptr + k);
    b0.q[1] = *(const uint4*)(b_ptr + k + 16);
    b1.q[0] = *(const uint4*)(b_ptr + bstep + k);
    b1.q[1] = *(const uint4*)(b_ptr + bstep + k + 16);
    b2.q[0] = *(const uint4*)(b_ptr + 2 * bstep + k);
    b2.q[1] = *(const uint4*)(b_ptr + 2 * bstep + k + 16);
    b3.q[0] = *(const uint4*)(b_ptr + 3 * bstep + k);
    b3.q[1] = *(const uint4*)(b_ptr + 3 * bstep + k + 16);
    acc0 = __builtin_amdgcn_wmma_f32_16x16x32_bf16(false, a.v, false, b0.v, (short)0, acc0, false, false);
    acc1 = __builtin_amdgcn_wmma_f32_16x16x32_bf16(false, a.v, false, b1.v, (short)0, acc1, false, false);
    acc2 = __builtin_amdgcn_wmma_f32_16x16x32_bf16(false, a.v, false, b2.v, (short)0, acc2, false, false);
    acc3 = __builtin_amdgcn_wmma_f32_16x16x32_bf16(false, a.v, false, b3.v, (short)0, acc3, false, false);
  }
  // C/D layout: lanes<16 hold M=r, lanes>=16 hold M=8+r; N = lane%16
  float* c_ptr = C + (size_t)(mt * 16 + (lane >> 4) * 8) * Nc + ng * 64 + lmod;
  #pragma unroll
  for (int r = 0; r < 8; ++r) {
    float* row = c_ptr + (size_t)r * Nc;
    row[0]  = acc0[r];
    row[16] = acc1[r];
    row[32] = acc2[r];
    row[48] = acc3[r];
  }
}

// ---------------- attention coefficients el/er : [N, HEADS] ----------------
// one wave per (node, head); lanes span outdim, shuffle-reduce.
__global__ void k_attn(const float* __restrict__ feat, const float* __restrict__ al,
                       const float* __restrict__ ar, float* __restrict__ el,
                       float* __restrict__ er, int n, int outdim) {
  int wid  = (blockIdx.x * blockDim.x + threadIdx.x) >> 5;
  int lane = threadIdx.x & 31;
  if (wid >= n * HEADS) return;
  int node = wid >> 2, head = wid & 3;   // HEADS == 4
  const float* f  = feat + (size_t)node * HEADS * outdim + head * outdim;
  const float* pl = al + head * outdim;
  const float* pr = ar + head * outdim;
  float sl = 0.f, sr = 0.f;
  for (int o = lane; o < outdim; o += 32) { float v = f[o]; sl += v * pl[o]; sr += v * pr[o]; }
  #pragma unroll
  for (int k = 16; k; k >>= 1) { sl += __shfl_xor(sl, k, 32); sr += __shfl_xor(sr, k, 32); }
  if (lane == 0) { el[wid] = sl; er[wid] = sr; }
}

// ---------------- init emax/-inf, denom/0, rst/0 ----------------
__global__ void k_init(float* __restrict__ emax, float* __restrict__ denom,
                       float* __restrict__ rst, int nh, int nrst) {
  int i = blockIdx.x * blockDim.x + threadIdx.x;
  if (i < nh) { emax[i] = -__builtin_inff(); denom[i] = 0.f; }
  if (i < nrst) rst[i] = 0.f;
}

// ---------------- pass 1: segment max of leaky_relu(el[src]+er[dst]) ----------------
__global__ void k_edge_max(const int* __restrict__ src, const int* __restrict__ dst,
                           const float* __restrict__ el, const float* __restrict__ er,
                           float* __restrict__ emax, int ne) {
  int i = blockIdx.x * blockDim.x + threadIdx.x;
  if (i >= ne * HEADS) return;
  int e = i >> 2, h = i & 3;
  int s = src[e], d = dst[e];
  float v = el[s * HEADS + h] + er[d * HEADS + h];
  v = v > 0.f ? v : NEG_SLOPE * v;
  atomicMaxF(&emax[d * HEADS + h], v);
}

// ---------------- pass 2: ee = exp(e - emax[dst]); denom += ee ----------------
__global__ void k_edge_exp(const int* __restrict__ src, const int* __restrict__ dst,
                           const float* __restrict__ el, const float* __restrict__ er,
                           const float* __restrict__ emax, float* __restrict__ denom,
                           float* __restrict__ ee, int ne) {
  int i = blockIdx.x * blockDim.x + threadIdx.x;
  if (i >= ne * HEADS) return;
  int e = i >> 2, h = i & 3;
  int s = src[e], d = dst[e];
  float v = el[s * HEADS + h] + er[d * HEADS + h];
  v = v > 0.f ? v : NEG_SLOPE * v;
  float x = __expf(v - emax[d * HEADS + h]);
  ee[i] = x;
  atomicAdd(&denom[d * HEADS + h], x);
}

// ---------------- pass 3 (layers 1-3, outdim==64): rst[dst] += alpha*feat[src] ----
// one wave per (edge, head); lanes span the 64-float head slice so every
// gather / atomic instruction touches 32 consecutive addresses.
__global__ void k_edge_aggr64(const int* __restrict__ src, const int* __restrict__ dst,
                              const float* __restrict__ ee, const float* __restrict__ denom,
                              const float* __restrict__ feat, float* __restrict__ rst,
                              int ne) {
  int wid  = (blockIdx.x * blockDim.x + threadIdx.x) >> 5;
  int lane = threadIdx.x & 31;
  if (wid >= ne * HEADS) return;
  int e = wid >> 2, h = wid & 3;
  int s = src[e], d = dst[e];
  float alpha = ee[wid] / denom[d * HEADS + h];
  const float* f = feat + (size_t)s * HO3 + h * HID;
  float*       r = rst  + (size_t)d * HO3 + h * HID;
  atomicAdd(&r[lane],      alpha * f[lane]);
  atomicAdd(&r[lane + 32], alpha * f[lane + 32]);
}

// ---------------- pass 3 (layer 4, outdim==2): thread per (edge, head) ----------
__global__ void k_edge_aggr_small(const int* __restrict__ src, const int* __restrict__ dst,
                                  const float* __restrict__ ee, const float* __restrict__ denom,
                                  const float* __restrict__ feat, float* __restrict__ rst,
                                  int ne, int outdim) {
  int i = blockIdx.x * blockDim.x + threadIdx.x;
  if (i >= ne * HEADS) return;
  int e = i >> 2, h = i & 3;
  int s = src[e], d = dst[e];
  float alpha = ee[i] / denom[d * HEADS + h];
  const float* f = feat + (size_t)s * HEADS * outdim + h * outdim;
  float*       r = rst  + (size_t)d * HEADS * outdim + h * outdim;
  for (int o = 0; o < outdim; ++o) atomicAdd(&r[o], alpha * f[o]);
}

// ---------------- residual + bias + ELU; write fp32 h and bf16 h ----------------
__global__ void k_finalize(const float* __restrict__ rst, const float* __restrict__ bias,
                           float* __restrict__ h, unsigned short* __restrict__ hb,
                           int n, int ho, int residual) {
  int i = blockIdx.x * blockDim.x + threadIdx.x;
  if (i >= n * ho) return;
  int j = i % ho;
  float v = rst[i] + bias[j] + (residual ? h[i] : 0.f);
  v = v > 0.f ? v : (__expf(v) - 1.f);
  h[i] = v;
  hb[i] = f2bf(v);
}

// ---------------- layer-4 projections: feat4 = h*W4^T, res4 = h*resW4^T ----------------
__global__ void k_gemm4(const float* __restrict__ h, const float* __restrict__ W4,
                        const float* __restrict__ resW4, float* __restrict__ feat4,
                        float* __restrict__ res4, int n) {
  int i = blockIdx.x * blockDim.x + threadIdx.x;
  if (i >= n * HO4) return;
  int node = i / HO4, j = i - node * HO4;
  const float* hr = h + (size_t)node * HO3;
  const float* w  = W4 + j * HO3;
  const float* rw = resW4 + j * HO3;
  float s = 0.f, r = 0.f;
  for (int k = 0; k < HO3; ++k) { float hv = hr[k]; s += hv * w[k]; r += hv * rw[k]; }
  feat4[i] = s; res4[i] = r;
}

// ---------------- final: + res + bias, softmax over C per head, mean over heads ----------------
__global__ void k_out(const float* __restrict__ rst4, const float* __restrict__ res4,
                      const float* __restrict__ b4, float* __restrict__ out, int n) {
  int i = blockIdx.x * blockDim.x + threadIdx.x;
  if (i >= n) return;
  float a0 = 0.f, a1 = 0.f;
  #pragma unroll
  for (int h = 0; h < HEADS; ++h) {
    float v0 = rst4[i * HO4 + h * CLS]     + res4[i * HO4 + h * CLS]     + b4[h * CLS];
    float v1 = rst4[i * HO4 + h * CLS + 1] + res4[i * HO4 + h * CLS + 1] + b4[h * CLS + 1];
    float m  = fmaxf(v0, v1);
    float e0 = __expf(v0 - m), e1 = __expf(v1 - m);
    float s  = e0 + e1;
    a0 += e0 / s; a1 += e1 / s;
  }
  out[i * 2]     = a0 * 0.25f;
  out[i * 2 + 1] = a1 * 0.25f;
}

static inline int cdiv(long a, long b) { return (int)((a + b - 1) / b); }

extern "C" void kernel_launch(void* const* d_in, const int* in_sizes, int n_in,
                              void* d_out, int out_size, void* d_ws, size_t ws_size,
                              hipStream_t stream) {
  const float* x     = (const float*)d_in[0];
  const int*   src   = (const int*)  d_in[1];
  const int*   dst   = (const int*)  d_in[2];
  const float* W[4]  = { (const float*)d_in[3], (const float*)d_in[7],
                         (const float*)d_in[11], (const float*)d_in[15] };
  const float* al[4] = { (const float*)d_in[4], (const float*)d_in[8],
                         (const float*)d_in[12], (const float*)d_in[16] };
  const float* ar[4] = { (const float*)d_in[5], (const float*)d_in[9],
                         (const float*)d_in[13], (const float*)d_in[17] };
  const float* bb[4] = { (const float*)d_in[6], (const float*)d_in[10],
                         (const float*)d_in[14], (const float*)d_in[18] };
  const float* resW4 = (const float*)d_in[19];
  float* out = (float*)d_out;

  // ---- carve workspace ----
  char* p = (char*)d_ws;
  auto alloc = [&](size_t bytes) -> void* {
    void* r = (void*)p; p += (bytes + 255) & ~(size_t)255; return r;
  };
  unsigned short* hb   = (unsigned short*)alloc((size_t)NN * HO3 * 2);  // bf16 activations
  unsigned short* wb   = (unsigned short*)alloc((size_t)HO3 * HO3 * 2); // bf16 weights
  float* feat  = (float*)alloc((size_t)NN * HO3 * 4);
  float* hbuf  = (float*)alloc((size_t)NN * HO3 * 4);
  float* rst   = (float*)alloc((size_t)NN * HO3 * 4);
  float* el    = (float*)alloc((size_t)NN * HEADS * 4);
  float* er    = (float*)alloc((size_t)NN * HEADS * 4);
  float* emax  = (float*)alloc((size_t)NN * HEADS * 4);
  float* denom = (float*)alloc((size_t)NN * HEADS * 4);
  float* ee    = (float*)alloc((size_t)NE * HEADS * 4);
  float* res4  = (float*)alloc((size_t)NN * HO4 * 4);

  const int B = 256;
  const int gNHw  = cdiv((long)NN * HEADS * 32, B);   // wave per (node,head)
  const int gEH   = cdiv((long)NE * HEADS, B);        // thread per (edge,head)
  const int gEHw  = cdiv((long)NE * HEADS * 32, B);   // wave per (edge,head)
  const int gNHO  = cdiv((long)NN * HO3, B);
  const int gGemm = cdiv((long)(NN / 16) * (HO3 / 64), 8); // 8 waves/block, 16x64 tiles

  // ================= layers 1..3 (WMMA GEMM + edge softmax aggregate) =================
  for (int L = 0; L < 3; ++L) {
    const int K = (L == 0) ? IND : HO3;
    if (L == 0) {
      k_f2bf<<<cdiv((long)NN * IND, B), B, 0, stream>>>(x, hb, NN * IND);
    } // for L>0, hb already holds bf16 of previous layer's output (from k_finalize)
    k_f2bf<<<cdiv((long)HO3 * K, B), B, 0, stream>>>(W[L], wb, HO3 * K);

    k_gemm_bf16_nt<<<gGemm, B, 0, stream>>>(hb, wb, feat, NN, HO3, K);
    k_attn<<<gNHw, B, 0, stream>>>(feat, al[L], ar[L], el, er, NN, HID);
    k_init<<<gNHO, B, 0, stream>>>(emax, denom, rst, NN * HEADS, NN * HO3);
    k_edge_max <<<gEH, B, 0, stream>>>(src, dst, el, er, emax, NE);
    k_edge_exp <<<gEH, B, 0, stream>>>(src, dst, el, er, emax, denom, ee, NE);
    k_edge_aggr64<<<gEHw, B, 0, stream>>>(src, dst, ee, denom, feat, rst, NE);
    k_finalize <<<gNHO, B, 0, stream>>>(rst, bb[L], hbuf, hb, NN, HO3, L > 0 ? 1 : 0);
  }

  // ================= layer 4 (skinny: scalar fp32, fused resW4) =================
  k_gemm4<<<cdiv((long)NN * HO4, B), B, 0, stream>>>(hbuf, W[3], resW4, feat, res4, NN);
  k_attn<<<gNHw, B, 0, stream>>>(feat, al[3], ar[3], el, er, NN, CLS);
  k_init<<<cdiv((long)NN * HO4, B), B, 0, stream>>>(emax, denom, rst, NN * HEADS, NN * HO4);
  k_edge_max <<<gEH, B, 0, stream>>>(src, dst, el, er, emax, NE);
  k_edge_exp <<<gEH, B, 0, stream>>>(src, dst, el, er, emax, denom, ee, NE);
  k_edge_aggr_small<<<gEH, B, 0, stream>>>(src, dst, ee, denom, feat, rst, NE, CLS);
  k_out<<<cdiv(NN, B), B, 0, stream>>>(rst, res4, bb[3], out, NN);
}
